// SparseMoE_55070070669772
// MI455X (gfx1250) — matrix operand order repaired and checked by reading, not compiled, not run on hardware
//
#include <hip/hip_runtime.h>
#include <stdint.h>

#define D_MODEL    1024
#define EXPERT_D   4096
#define NUM_EXPERTS   8
#define T_TOK      8192
#define HC          128   // h-chunk per block iteration

typedef __attribute__((ext_vector_type(16))) __bf16 v16bf;
typedef __attribute__((ext_vector_type(8)))  float  v8f;

union FragU { v16bf v; uint4 u[2]; };

static __device__ __forceinline__ unsigned short f32_to_bf16(float f) {
  unsigned int u = __float_as_uint(f);
  unsigned int r = u + 0x7FFFu + ((u >> 16) & 1u);   // round-to-nearest-even
  return (unsigned short)(r >> 16);
}

// ---------------------------------------------------------------- init
__global__ void moe_init_kernel(int* cnt, float* meanp) {
  int i = threadIdx.x;
  if (i < NUM_EXPERTS) { cnt[i] = 0; meanp[i] = 0.0f; }
}

// ---------------------------------------------------------------- router
__global__ __launch_bounds__(256) void moe_router_kernel(
    const float* __restrict__ x, const float* __restrict__ Wr,
    const float* __restrict__ br,
    int* __restrict__ cnt, float* __restrict__ meanp,
    int* __restrict__ etok, float* __restrict__ ewt)
{
  __shared__ __align__(16) float sWr[NUM_EXPERTS * D_MODEL];
  const int tid = threadIdx.x;
  for (int i = tid; i < NUM_EXPERTS * D_MODEL; i += 256) sWr[i] = Wr[i];
  __syncthreads();

  const int t = blockIdx.x * 256 + tid;
  float acc[NUM_EXPERTS];
#pragma unroll
  for (int e = 0; e < NUM_EXPERTS; ++e) acc[e] = br[e];

  const float4* xv = (const float4*)(x + (size_t)t * D_MODEL);
  for (int i = 0; i < D_MODEL / 4; ++i) {
    float4 v = xv[i];
#pragma unroll
    for (int e = 0; e < NUM_EXPERTS; ++e) {
      float4 w = *(const float4*)(sWr + e * D_MODEL + i * 4);
      acc[e] += v.x * w.x + v.y * w.y + v.z * w.z + v.w * w.w;
    }
  }
  // softmax over 8
  float mx = acc[0];
#pragma unroll
  for (int e = 1; e < NUM_EXPERTS; ++e) mx = fmaxf(mx, acc[e]);
  float p[NUM_EXPERTS], s = 0.0f;
#pragma unroll
  for (int e = 0; e < NUM_EXPERTS; ++e) { p[e] = __expf(acc[e] - mx); s += p[e]; }
  const float inv = 1.0f / s;
#pragma unroll
  for (int e = 0; e < NUM_EXPERTS; ++e) { p[e] *= inv; atomicAdd(&meanp[e], p[e]); }
  // top-2 (first max wins ties, matching top_k)
  int e1 = 0; float p1 = p[0];
#pragma unroll
  for (int e = 1; e < NUM_EXPERTS; ++e) if (p[e] > p1) { p1 = p[e]; e1 = e; }
  int e2 = (e1 == 0) ? 1 : 0; float p2 = p[e2];
#pragma unroll
  for (int e = 0; e < NUM_EXPERTS; ++e)
    if (e != e1 && p[e] > p2) { p2 = p[e]; e2 = e; }
  const float denom = p1 + p2;
  const float w1 = p1 / denom, w2 = p2 / denom;

  int pos1 = atomicAdd(&cnt[e1], 1);
  etok[e1 * T_TOK + pos1] = t; ewt[e1 * T_TOK + pos1] = w1;
  int pos2 = atomicAdd(&cnt[e2], 1);
  etok[e2 * T_TOK + pos2] = t; ewt[e2 * T_TOK + pos2] = w2;
}

// ---------------------------------------------------------------- aux loss
__global__ void moe_aux_kernel(const float* __restrict__ meanp,
                               float* __restrict__ out_aux) {
  if (threadIdx.x == 0 && blockIdx.x == 0) {
    float a = 0.0f;
    for (int e = 0; e < NUM_EXPERTS; ++e) {
      float mp = meanp[e] * (1.0f / (float)T_TOK);
      float d = mp - 1.0f / (float)NUM_EXPERTS;
      a += d * d;
    }
    out_aux[0] = a;
  }
}

// ------------------------------------------------- weight pack (f32 -> bf16, WMMA-B-fragment order)
// W1p fragment = 16 h-rows x 32 d-cols = 512 bf16 = 1KB, index ((e*256+th)*32+kk)
// within fragment lane l owns 16 elems: B[k = (j<8 ? (l>>4)*8 + j : 16 + (l>>4)*8 + j-8)][n = l&15]
__global__ __launch_bounds__(256) void moe_pack_w1_kernel(
    const float* __restrict__ W1, unsigned short* __restrict__ W1p)
{
  size_t idx = (size_t)blockIdx.x * 256 + threadIdx.x;   // 33,554,432 total
  const int j    = (int)(idx & 15);
  const int lane = (int)((idx >> 4) & 31);
  const size_t frag = idx >> 9;
  const int kk = (int)(frag & 31);
  const int th = (int)((frag >> 5) & 255);
  const int e  = (int)(frag >> 13);
  const int row = th * 16 + (lane & 15);
  const int col = kk * 32 + (lane >> 4) * 8 + (j & 7) + (j >> 3) * 16;
  W1p[idx] = f32_to_bf16(W1[((size_t)e * EXPERT_D + row) * D_MODEL + col]);
}

// W2p fragment = 16 d-rows x 32 h-cols, index ((e*64+td)*128+hh)
__global__ __launch_bounds__(256) void moe_pack_w2_kernel(
    const float* __restrict__ W2, unsigned short* __restrict__ W2p)
{
  size_t idx = (size_t)blockIdx.x * 256 + threadIdx.x;   // 33,554,432 total
  const int j    = (int)(idx & 15);
  const int lane = (int)((idx >> 4) & 31);
  const size_t frag = idx >> 9;
  const int hh = (int)(frag & 127);
  const int td = (int)((frag >> 7) & 63);
  const int e  = (int)(frag >> 13);
  const int row = td * 16 + (lane & 15);
  const int col = hh * 32 + (lane >> 4) * 8 + (j & 7) + (j >> 3) * 16;
  W2p[idx] = f32_to_bf16(W2[((size_t)e * D_MODEL + row) * EXPERT_D + col]);
}

// ---------------------------------------------------------------- zero output
__global__ void moe_zero_out_kernel(float* __restrict__ out) {
  size_t i = (size_t)blockIdx.x * 256 + threadIdx.x;
  out[i] = 0.0f;
}

// ---------------------------------------------------------------- fused expert FFN (WMMA bf16)
__global__ __launch_bounds__(256) void moe_ffn_kernel(
    const float* __restrict__ x,
    const unsigned short* __restrict__ W1p,
    const unsigned short* __restrict__ W2p,
    const float* __restrict__ b1,
    const float* __restrict__ b2,
    const int* __restrict__ cnt,
    const int* __restrict__ etok,
    const float* __restrict__ ewt,
    float* __restrict__ out)
{
  const int e    = blockIdx.y;
  const int base = blockIdx.x * 16;
  const int c_e  = cnt[e];
  if (base >= c_e) return;

  __shared__ __align__(16) unsigned short lds_x[16 * D_MODEL];  // 32KB bf16 token tile
  __shared__ __align__(16) unsigned short lds_h[16 * HC];       // 4KB bf16 hidden chunk
  __shared__ int   lds_tok[16];
  __shared__ float lds_wt[16];

  const int tid  = threadIdx.x;
  const int lane = tid & 31;
  const int wv   = tid >> 5;       // wave 0..7
  const int n    = lane & 15;      // column / row-in-halfwave
  const int kh   = lane >> 4;      // K-half selector

  if (tid < 16) {
    int r = base + tid;
    if (r < c_e) { lds_tok[tid] = etok[e * T_TOK + r]; lds_wt[tid] = ewt[e * T_TOK + r]; }
    else         { lds_tok[tid] = 0;                   lds_wt[tid] = 0.0f; }
  }
  __syncthreads();

  // gather + convert token tile to bf16 in LDS
  for (int i = tid; i < 16 * D_MODEL; i += 256) {
    int row = i >> 10;
    int col = i & (D_MODEL - 1);
    lds_x[i] = f32_to_bf16(x[(size_t)lds_tok[row] * D_MODEL + col]);
  }
  __syncthreads();

  v8f acc2[8];
#pragma unroll
  for (int i = 0; i < 8; ++i) acc2[i] = (v8f){0,0,0,0,0,0,0,0};

  // A-fragment base for stage 1 (rows = tokens, K over d)
  const unsigned short* a1p = lds_x + n * D_MODEL + kh * 8;

  for (int c = 0; c < EXPERT_D / HC; ++c) {            // 32 h-chunks
    // ---- stage 1: hidden[16, 16] (this wave's h-tile), K = 1024
    v8f acc1 = (v8f){0,0,0,0,0,0,0,0};
    const int th = c * (HC / 16) + wv;                 // global 16-row h-tile
    const unsigned short* b1p =
        W1p + (((size_t)e * (EXPERT_D / 16) + th) * (D_MODEL / 32)) * 512 + lane * 16;
#pragma unroll 4
    for (int kk = 0; kk < D_MODEL / 32; ++kk) {
      FragU A, B;
      A.u[0] = *(const uint4*)(a1p + kk * 32);         // K = kk*32 + kh*8 .. +7
      A.u[1] = *(const uint4*)(a1p + kk * 32 + 16);    // K = kk*32 + 16 + kh*8 ..
      B.u[0] = *(const uint4*)(b1p + (size_t)kk * 512);
      B.u[1] = *(const uint4*)(b1p + (size_t)kk * 512 + 8);
      acc1 = __builtin_amdgcn_wmma_f32_16x16x32_bf16(false, A.v, false, B.v,
                                                     (short)0, acc1, false, false);
    }
    // bias + relu + store bf16 hidden to LDS (C layout: M = kh*8+r, N = n)
    const float b1v = b1[(size_t)e * EXPERT_D + c * HC + wv * 16 + n];
#pragma unroll
    for (int r = 0; r < 8; ++r) {
      float v = acc1[r] + b1v;
      v = v > 0.0f ? v : 0.0f;
      lds_h[(kh * 8 + r) * HC + wv * 16 + n] = f32_to_bf16(v);
    }
    __syncthreads();

    // ---- stage 2: y[16, 128 cols of this wave] += hidden_chunk @ W2^T
    const unsigned short* a2p = lds_h + n * HC + kh * 8;
#pragma unroll
    for (int hh = 0; hh < HC / 32; ++hh) {             // 4 K-steps
      FragU A2;
      A2.u[0] = *(const uint4*)(a2p + hh * 32);
      A2.u[1] = *(const uint4*)(a2p + hh * 32 + 16);
      const int hhg = c * (HC / 32) + hh;              // global h K-step 0..127
#pragma unroll
      for (int td = 0; td < 8; ++td) {
        const int tdg = wv * 8 + td;                   // global 16-col d-tile
        const unsigned short* b2p =
            W2p + (((size_t)e * (D_MODEL / 16) + tdg) * (EXPERT_D / 32) + hhg) * 512
                + lane * 16;
        FragU B2;
        B2.u[0] = *(const uint4*)(b2p);
        B2.u[1] = *(const uint4*)(b2p + 8);
        acc2[td] = __builtin_amdgcn_wmma_f32_16x16x32_bf16(false, A2.v, false, B2.v,
                                                           (short)0, acc2[td], false, false);
      }
    }
    __syncthreads();
  }

  // ---- epilogue: out[tok] += wt * (y + b2)
#pragma unroll
  for (int td = 0; td < 8; ++td) {
    const int dcol = (wv * 8 + td) * 16 + n;
    const float b2v = b2[(size_t)e * D_MODEL + dcol];
#pragma unroll
    for (int r = 0; r < 8; ++r) {
      const int m = kh * 8 + r;
      const float yv = lds_wt[m] * (acc2[td][r] + b2v);
      atomicAdd(&out[(size_t)lds_tok[m] * D_MODEL + dcol], yv);
    }
  }
}

// ---------------------------------------------------------------- launch
extern "C" void kernel_launch(void* const* d_in, const int* in_sizes, int n_in,
                              void* d_out, int out_size, void* d_ws, size_t ws_size,
                              hipStream_t stream) {
  (void)in_sizes; (void)n_in; (void)out_size; (void)ws_size;
  const float* x  = (const float*)d_in[0];
  const float* Wr = (const float*)d_in[1];
  const float* br = (const float*)d_in[2];
  const float* W1 = (const float*)d_in[3];
  const float* b1 = (const float*)d_in[4];
  const float* W2 = (const float*)d_in[5];
  const float* b2 = (const float*)d_in[6];
  float* out = (float*)d_out;

  char* ws = (char*)d_ws;
  int*   cnt   = (int*)ws;                                  // 8 ints
  float* meanp = (float*)(ws + 64);                         // 8 floats
  int*   etok  = (int*)(ws + 1024);                         // 8*8192 ints
  float* ewt   = (float*)(ws + 1024 + NUM_EXPERTS * T_TOK * 4);
  unsigned short* W1p = (unsigned short*)(ws + (1 << 20));  // 64 MB bf16 packed
  unsigned short* W2p = W1p + (size_t)NUM_EXPERTS * EXPERT_D * D_MODEL;

  const size_t NW = (size_t)NUM_EXPERTS * EXPERT_D * D_MODEL;  // 33,554,432

  moe_init_kernel<<<1, 32, 0, stream>>>(cnt, meanp);
  moe_router_kernel<<<T_TOK / 256, 256, 0, stream>>>(x, Wr, br, cnt, meanp, etok, ewt);
  moe_aux_kernel<<<1, 32, 0, stream>>>(meanp, out + (size_t)T_TOK * D_MODEL);
  moe_pack_w1_kernel<<<(unsigned)(NW / 256), 256, 0, stream>>>(W1, W1p);
  moe_pack_w2_kernel<<<(unsigned)(NW / 256), 256, 0, stream>>>(W2, W2p);
  moe_zero_out_kernel<<<(T_TOK * D_MODEL) / 256, 256, 0, stream>>>(out);
  dim3 grid(T_TOK / 16, NUM_EXPERTS);
  moe_ffn_kernel<<<grid, 256, 0, stream>>>(x, W1p, W2p, b1, b2, cnt, etok, ewt, out);
}
